// CausalMultiHeadSelfAttention_6090263626362
// MI455X (gfx1250) — matrix operand order repaired and checked by reading, not compiled, not run on hardware
//
#include <hip/hip_runtime.h>

typedef __attribute__((ext_vector_type(16))) __bf16        bf16x16;
typedef __attribute__((ext_vector_type(8)))  float         f32x8;
typedef __attribute__((ext_vector_type(8)))  unsigned int  u32x8;

#define DM   1024
#define SEQ  2048
#define NH   16
#define DK   64

#if defined(__HIP_DEVICE_COMPILE__) && __has_builtin(__builtin_amdgcn_global_load_async_to_lds_b128)
#define HAVE_ASYNC_LDS 1
#else
#define HAVE_ASYNC_LDS 0
#endif

#if HAVE_ASYNC_LDS
// 16-byte async copy global -> LDS (GLOBAL_LOAD_ASYNC_TO_LDS_B128, ASYNCcnt).
// Builtin params (from hipcc diagnostic): v4i addrspace(1)* src, v4i addrspace(3)* dst,
// imm offset, imm cpol. LDS pointer produced via integer round-trip; LLVM folds it
// back to the addrspace(3) object.
typedef int v4i_vs __attribute__((__vector_size__(4 * sizeof(int))));
typedef __attribute__((address_space(1))) v4i_vs* gptr128_t;
typedef __attribute__((address_space(3))) v4i_vs* lptr128_t;
__device__ __forceinline__ void cp16_async(void* lds, const void* g) {
    __builtin_amdgcn_global_load_async_to_lds_b128(
        (gptr128_t)(unsigned long long)g,
        (lptr128_t)(unsigned)(unsigned long long)lds,
        0, 0);
}
__device__ __forceinline__ void wait_async() {
#if __has_builtin(__builtin_amdgcn_s_wait_asynccnt)
    __builtin_amdgcn_s_wait_asynccnt(0);
#else
    asm volatile("s_wait_asynccnt 0x0" ::: "memory");
#endif
}
#else
__device__ __forceinline__ void cp16_async(void* lds, const void* g) {
    *(uint4*)lds = *(const uint4*)g;
}
__device__ __forceinline__ void wait_async() {}
#endif

__device__ __forceinline__ unsigned short f2bf(float f) {
    unsigned int u = __builtin_bit_cast(unsigned int, f);
    u += 0x7fffu + ((u >> 16) & 1u);            // round-to-nearest-even
    return (unsigned short)(u >> 16);
}

// A-operand fragment (16 rows x 32 K, bf16), row-major LDS tile.
// Lanes 0-15: M=lane, K = 0..7 & 16..23 ; lanes 16-31: M=lane-16, K = 8..15 & 24..31.
__device__ __forceinline__ bf16x16 ld_a_frag(const unsigned short* base, int strideE) {
    int lane = threadIdx.x & 31;
    const unsigned short* p = base + (lane & 15) * strideE + ((lane >> 4) << 3);
    uint4 lo = *(const uint4*)p;
    uint4 hi = *(const uint4*)(p + 16);
    u32x8 v = { lo.x, lo.y, lo.z, lo.w, hi.x, hi.y, hi.z, hi.w };
    return __builtin_bit_cast(bf16x16, v);
}

// B-operand fragment (32 K x 16 cols, bf16) from LDS tile stored [col][K] row-major.
// Lanes 0-15: N=lane, K = 0..7 & 16..23 ; lanes 16-31: N=lane-16, K = 8..15 & 24..31.
__device__ __forceinline__ bf16x16 ld_b_frag(const unsigned short* base, int strideE) {
    int lane = threadIdx.x & 31;
    const unsigned short* p = base + (lane & 15) * strideE + ((lane >> 4) << 3);
    uint4 lo = *(const uint4*)p;
    uint4 hi = *(const uint4*)(p + 16);
    u32x8 v = { lo.x, lo.y, lo.z, lo.w, hi.x, hi.y, hi.z, hi.w };
    return __builtin_bit_cast(bf16x16, v);
}

__device__ __forceinline__ f32x8 wmma_bf16(bf16x16 a, bf16x16 b, f32x8 c) {
    return __builtin_amdgcn_wmma_f32_16x16x32_bf16(false, a, false, b, (short)0, c, false, false);
}

// out[m,n] = sum_k A[m,k] * W[n,k]   (A: 4096x1024 f32, W: 1024x1024 f32)
// mode 0: store f32 at outF[m*1024+n]
// mode 1: RoPE, store bf16 at outB[((b*16+h)*2048+s)*64+d]      (Q/K, head-major)
// mode 2: no RoPE, store bf16 TRANSPOSED at outB[((b*16+h)*64+d)*2048+s]  (V, [d][s])
__global__ __launch_bounds__(256) void proj_gemm(const float* __restrict__ A,
        const float* __restrict__ W, float* __restrict__ outF,
        unsigned short* __restrict__ outB, const int* __restrict__ tokpos, int mode)
{
    __shared__ __align__(16) unsigned short As[64 * 32];     // 4 KB
    __shared__ __align__(16) unsigned short Bs[128 * 32];    // 8 KB
    int tid = threadIdx.x, lane = tid & 31, wave = tid >> 5;
    int bm = blockIdx.y * 64, bn = blockIdx.x * 128;
    int wm = (wave & 1) * 32, wn = (wave >> 1) * 32;
    f32x8 zero = {0.f,0.f,0.f,0.f,0.f,0.f,0.f,0.f};
    f32x8 acc0 = zero, acc1 = zero, acc2 = zero, acc3 = zero;

    for (int kt = 0; kt < DM; kt += 32) {
        __syncthreads();
        {   // A tile: 64 rows x 32 cols, f32 -> bf16
            int row = tid >> 2, cs = (tid & 3) * 8;
            const float* s = A + (size_t)(bm + row) * DM + kt + cs;
            float4 f0 = *(const float4*)s, f1 = *(const float4*)(s + 4);
            unsigned short* d = &As[row * 32 + cs];
            d[0]=f2bf(f0.x); d[1]=f2bf(f0.y); d[2]=f2bf(f0.z); d[3]=f2bf(f0.w);
            d[4]=f2bf(f1.x); d[5]=f2bf(f1.y); d[6]=f2bf(f1.z); d[7]=f2bf(f1.w);
        }
        {   // W tile: 128 rows (n) x 32 cols (k)
            int row = tid >> 1, cs = (tid & 1) * 16;
            const float* s = W + (size_t)(bn + row) * DM + kt + cs;
            unsigned short* d = &Bs[row * 32 + cs];
            #pragma unroll
            for (int j = 0; j < 4; ++j) {
                float4 f = *(const float4*)(s + j * 4);
                d[j*4+0]=f2bf(f.x); d[j*4+1]=f2bf(f.y);
                d[j*4+2]=f2bf(f.z); d[j*4+3]=f2bf(f.w);
            }
        }
        __syncthreads();
        bf16x16 a0 = ld_a_frag(&As[wm * 32], 32);
        bf16x16 a1 = ld_a_frag(&As[(wm + 16) * 32], 32);
        bf16x16 b0 = ld_b_frag(&Bs[wn * 32], 32);
        bf16x16 b1 = ld_b_frag(&Bs[(wn + 16) * 32], 32);
        acc0 = wmma_bf16(a0, b0, acc0);
        acc1 = wmma_bf16(a0, b1, acc1);
        acc2 = wmma_bf16(a1, b0, acc2);
        acc3 = wmma_bf16(a1, b1, acc3);
    }

    int lane15 = lane & 15, half = lane >> 4;
    f32x8 accs[4] = { acc0, acc1, acc2, acc3 };
    #pragma unroll
    for (int ti = 0; ti < 2; ++ti)
    #pragma unroll
    for (int tj = 0; tj < 2; ++tj) {
        f32x8 c = accs[ti * 2 + tj];
        int ncol = bn + wn + tj * 16 + lane15;
        int h = ncol >> 6, d = ncol & (DK - 1);
        float freq = 0.f;
        if (mode == 1)   // loop-invariant RoPE frequency for this column
            freq = __powf(10000.0f, -(float)(d & ~1) * (1.0f / 64.0f));
        #pragma unroll
        for (int r = 0; r < 8; ++r) {
            int mrow = bm + wm + ti * 16 + half * 8 + r;
            float v = c[r];
            if (mode == 0) {
                outF[(size_t)mrow * DM + ncol] = v;
            } else {
                float partner = __shfl_xor(v, 1, 32);   // adjacent column lives in lane^1
                int b = mrow >> 11, s = mrow & (SEQ - 1);
                float res = v;
                if (mode == 1) {
                    float ang = (float)tokpos[s] * freq;
                    float sn, cn; __sincosf(ang, &sn, &cn);
                    res = (d & 1) ? (partner * sn + v * cn) : (v * cn - partner * sn);
                }
                if (mode == 2)      // V: store [d][s] so attention streams it verbatim
                    outB[(((size_t)(b * NH + h)) * DK + d) * SEQ + s] = f2bf(res);
                else
                    outB[(((size_t)(b * NH + h)) * SEQ + s) * DK + d] = f2bf(res);
            }
        }
    }
}

// Flash attention, causal. Q,K bf16 (B,H,S,64); V bf16 transposed (B,H,64,S).
// Output f32 (B,S,1024). Double-buffered async K/V tile streaming.
__global__ __launch_bounds__(256) void attn_fwd(const unsigned short* __restrict__ Q,
        const unsigned short* __restrict__ K, const unsigned short* __restrict__ Vt,
        float* __restrict__ O)
{
    __shared__ __align__(16) unsigned short Qs[128 * 64];    // 16 KB
    __shared__ __align__(16) unsigned short Ks[2][32 * 64];  // 8 KB   [key][d]
    __shared__ __align__(16) unsigned short Vs[2][64 * 32];  // 8 KB   [d][key]
    __shared__ __align__(16) unsigned short Ps[8][16 * 32];  // 8 KB   per-wave P tiles
    int tid = threadIdx.x, lane = tid & 31, w = tid >> 5;
    int qt = blockIdx.x, bh = blockIdx.y;
    int b = bh >> 4, h = bh & 15;
    int lane15 = lane & 15, half = lane >> 4;

    const unsigned short* Kbase = K  + (size_t)bh * SEQ * DK;
    const unsigned short* Vbase = Vt + (size_t)bh * DK * SEQ;
    int vrow = tid >> 2, vcs = (tid & 3) * 8;

    // issue async loads for key-tile 0 into buffer 0
    cp16_async(&Ks[0][tid * 8], Kbase + tid * 8);
    cp16_async(&Vs[0][vrow * 32 + vcs], Vbase + (size_t)vrow * SEQ + vcs);

    size_t qbase = ((size_t)bh * SEQ + qt * 128) * DK;
    #pragma unroll
    for (int j = 0; j < 4; ++j)
        *(uint4*)&Qs[tid * 32 + j * 8] = *(const uint4*)&Q[qbase + tid * 32 + j * 8];

    wait_async();
    __syncthreads();

    int q0 = qt * 128 + w * 16;
    bf16x16 qa0 = ld_a_frag(&Qs[(w * 16) * 64], 64);
    bf16x16 qa1 = ld_a_frag(&Qs[(w * 16) * 64 + 32], 64);

    f32x8 zero = {0.f,0.f,0.f,0.f,0.f,0.f,0.f,0.f};
    f32x8 o0 = zero, o1 = zero, o2 = zero, o3 = zero;
    float mi[8], li[8];
    #pragma unroll
    for (int r = 0; r < 8; ++r) { mi[r] = -1e30f; li[r] = 0.f; }

    int ntiles = qt * 4 + 4;     // key tiles of 32 up to block's last query row
    for (int t = 0; t < ntiles; ++t) {
        int kb = t * 32, buf = t & 1;
        if (t + 1 < ntiles) {    // stream next tile while computing this one
            int nb = buf ^ 1, nkb = kb + 32;
            cp16_async(&Ks[nb][tid * 8], Kbase + (size_t)nkb * DK + tid * 8);
            cp16_async(&Vs[nb][vrow * 32 + vcs],
                       Vbase + (size_t)vrow * SEQ + nkb + vcs);
        }
        if (kb <= q0 + 15) {     // wave-uniform skip of fully-masked tiles
            const unsigned short* Kt = Ks[buf];
            const unsigned short* Vb = Vs[buf];
            f32x8 s0 = zero, s1 = zero;
            s0 = wmma_bf16(qa0, ld_b_frag(&Kt[0], 64), s0);
            s0 = wmma_bf16(qa1, ld_b_frag(&Kt[32], 64), s0);
            s1 = wmma_bf16(qa0, ld_b_frag(&Kt[16 * 64], 64), s1);
            s1 = wmma_bf16(qa1, ld_b_frag(&Kt[16 * 64 + 32], 64), s1);
            unsigned short* P = Ps[w];
            #pragma unroll
            for (int r = 0; r < 8; ++r) {
                int qg = q0 + half * 8 + r;
                int kg0 = kb + lane15, kg1 = kb + 16 + lane15;
                float e0 = (kg0 <= qg) ? s0[r] * 0.125f : -1e30f;
                float e1 = (kg1 <= qg) ? s1[r] * 0.125f : -1e30f;
                float tmax = fmaxf(e0, e1);
                #pragma unroll
                for (int msk = 8; msk >= 1; msk >>= 1)
                    tmax = fmaxf(tmax, __shfl_xor(tmax, msk, 32));
                float mnew = fmaxf(mi[r], tmax);
                float corr = __expf(mi[r] - mnew);
                float p0 = __expf(e0 - mnew), p1 = __expf(e1 - mnew);
                float rsum = p0 + p1;
                #pragma unroll
                for (int msk = 8; msk >= 1; msk >>= 1)
                    rsum += __shfl_xor(rsum, msk, 32);
                li[r] = li[r] * corr + rsum;
                mi[r] = mnew;
                o0[r] *= corr; o1[r] *= corr; o2[r] *= corr; o3[r] *= corr;
                int mr = half * 8 + r;
                P[mr * 32 + lane15] = f2bf(p0);
                P[mr * 32 + 16 + lane15] = f2bf(p1);
            }
            bf16x16 pa = ld_a_frag(P, 32);
            o0 = wmma_bf16(pa, ld_b_frag(&Vb[0], 32), o0);
            o1 = wmma_bf16(pa, ld_b_frag(&Vb[16 * 32], 32), o1);
            o2 = wmma_bf16(pa, ld_b_frag(&Vb[32 * 32], 32), o2);
            o3 = wmma_bf16(pa, ld_b_frag(&Vb[48 * 32], 32), o3);
        }
        wait_async();            // own next-tile loads complete
        __syncthreads();         // visible block-wide; safe to reuse buffers
    }
    #pragma unroll
    for (int r = 0; r < 8; ++r) {
        float inv = 1.0f / li[r];
        int s = q0 + half * 8 + r;
        float* dst = &O[((size_t)b * SEQ + s) * DM + h * DK];
        dst[0 * 16 + lane15] = o0[r] * inv;
        dst[1 * 16 + lane15] = o1[r] * inv;
        dst[2 * 16 + lane15] = o2[r] * inv;
        dst[3 * 16 + lane15] = o3[r] * inv;
    }
}

extern "C" void kernel_launch(void* const* d_in, const int* in_sizes, int n_in,
                              void* d_out, int out_size, void* d_ws, size_t ws_size,
                              hipStream_t stream) {
    const float* x  = (const float*)d_in[0];
    const int*   tp = (const int*)d_in[1];
    const float* Wq = (const float*)d_in[2];
    const float* Wk = (const float*)d_in[3];
    const float* Wv = (const float*)d_in[4];
    const float* Wo = (const float*)d_in[5];

    const size_t HE = (size_t)2 * NH * SEQ * DK;          // 4M elements
    unsigned short* Qb = (unsigned short*)d_ws;
    unsigned short* Kb = Qb + HE;
    unsigned short* Vb = Kb + HE;                         // stored [b,h][d][s]
    float* attn = (float*)(Vb + HE);                      // 4096x1024 f32

    dim3 blk(256);
    dim3 gProj(DM / 128, (2 * SEQ) / 64);                 // (8, 64)
    proj_gemm<<<gProj, blk, 0, stream>>>(x, Wq, nullptr, Qb, tp, 1);
    proj_gemm<<<gProj, blk, 0, stream>>>(x, Wk, nullptr, Kb, tp, 1);
    proj_gemm<<<gProj, blk, 0, stream>>>(x, Wv, nullptr, Vb, tp, 2);
    attn_fwd<<<dim3(SEQ / 128, 2 * NH), blk, 0, stream>>>(Qb, Kb, Vb, attn);
    proj_gemm<<<gProj, blk, 0, stream>>>(attn, Wo, (float*)d_out, nullptr, tp, 0);
}
